// DialogueTransformer_919123001547
// MI455X (gfx1250) — compile-verified, matrix operand order checked by reading, not run on hardware
//
#include <hip/hip_runtime.h>
#include <hip/hip_bf16.h>

// ---------------------------------------------------------------------------
// DialogueTransformer forward: B=8, S=2048, D=1024, H=8, HD=128
// 14x GEMM (16384 x 1024 x 1024): bf16 operands in HBM, TDM tensor_load_to_lds
// double-buffered staging, v_wmma_f32_16x16x32_bf16 compute (fp32 accum).
// + per-position 8x8 head-mixing attention (VALU).
// ---------------------------------------------------------------------------

typedef __bf16 bf16;
typedef bf16  v16bf  __attribute__((ext_vector_type(16)));
typedef bf16  v8bf   __attribute__((ext_vector_type(8)));
typedef bf16  bf16x4 __attribute__((ext_vector_type(4)));
typedef float v8f    __attribute__((ext_vector_type(8)));
typedef float f32x4  __attribute__((ext_vector_type(4)));
typedef unsigned int u32x4 __attribute__((ext_vector_type(4)));
typedef int   i32x4  __attribute__((ext_vector_type(4)));
typedef int   i32x8  __attribute__((ext_vector_type(8)));

namespace cfg {
constexpr int Bb = 8, S = 2048, D = 1024, H = 8, HD = 128;
constexpr int M = Bb * S;          // 16384 rows
constexpr int N = D, K = D;        // 1024
constexpr int BM = 128, BN = 128, BK = 64;
}

// ---------------------------------------------------------------------------
// TDM: issue tensor_load_to_lds of a 2D bf16 tile (rows x cols), row stride
// `stride` elements, into LDS at byte address lds_addr. Wave-level op
// (EXEC ignored); tracked by TENSORcnt.
// Toolchain uses the 6-arg builtin: (g0 u32x4, g1 i32x8, g2 i32x4, g3 i32x4,
// extra i32x8, cpol). Groups 2/3/extra unused for 2D tiles -> zero.
// ---------------------------------------------------------------------------
__device__ __forceinline__ void tdm_load_tile_bf16(const bf16* gptr,
                                                   unsigned lds_addr,
                                                   int rows, int cols, int stride)
{
    unsigned long long ga = (unsigned long long)gptr;

    u32x4 g0;
    g0[0] = 1u;                                    // count=1 valid, user mode
    g0[1] = lds_addr;                              // LDS byte address
    g0[2] = (unsigned)(ga & 0xFFFFFFFFu);          // global_addr[31:0]
    g0[3] = (unsigned)((ga >> 32) & 0x01FFFFFFu)   // global_addr[56:32]
          | 0x80000000u;                           // type=2 ("image")

    i32x8 g1;
    g1[0] = 0x00010000;                            // data_size=1 -> 2 bytes
    g1[1] = 0x00000000;                            // abar=0, tensor_dim0[15:0]=0
    g1[2] = 0x00000001;                            // tensor_dim0=65536 (no clamp)
    g1[3] = (cols << 16) | 0x1;                    // tile_dim0=cols, tensor_dim1=65536
    g1[4] = rows & 0xFFFF;                         // tile_dim1=rows, tile_dim2=0
    g1[5] = stride;                                // tensor_dim0_stride[31:0]
    g1[6] = 0;                                     // stride[47:32]=0, dim1_stride lo=0
    g1[7] = 0;

    i32x4 gz4 = (i32x4)(0);
    i32x8 gz8 = (i32x8)(0);
    __builtin_amdgcn_tensor_load_to_lds(g0, g1, gz4, gz4, gz8, 0);
}

// ---------------------------------------------------------------------------
// fp32 -> bf16 bulk convert (float4 in, bf16x4 out)
// ---------------------------------------------------------------------------
__launch_bounds__(256)
__global__ void f32_to_bf16_kernel(const float* __restrict__ in,
                                   bf16* __restrict__ out)
{
    long t = (long)blockIdx.x * 256 + threadIdx.x;
    f32x4 a = ((const f32x4*)in)[t];
    bf16x4 o = { (bf16)a[0], (bf16)a[1], (bf16)a[2], (bf16)a[3] };
    ((bf16x4*)out)[t] = o;
}

// ---------------------------------------------------------------------------
// x = input_seq + pe, output in bf16
// ---------------------------------------------------------------------------
__launch_bounds__(256)
__global__ void add_pe_kernel(const float* __restrict__ in,
                              const float* __restrict__ pe,
                              bf16* __restrict__ out)
{
    long t = (long)blockIdx.x * 256 + threadIdx.x;      // float4 index
    f32x4 a = ((const f32x4*)in)[t];
    f32x4 p = ((const f32x4*)pe)[t & (cfg::D / 4 - 1)];
    f32x4 s = a + p;
    bf16x4 o = { (bf16)s[0], (bf16)s[1], (bf16)s[2], (bf16)s[3] };
    ((bf16x4*)out)[t] = o;
}

// ---------------------------------------------------------------------------
// C[M,N] = A[M,K] @ W[N,K]^T + bias  (optional ReLU)
// bf16 operands in HBM, TDM double-buffered LDS staging, bf16 WMMA.
// Workgroup: 256 threads (8 waves), tile 128x128, K-stage 64.
// Wave grid 2(M) x 4(N): each wave owns 64x32 = 4x2 WMMA tiles.
// Output: fp32 (Cf) or bf16 (Cb) — exactly one is non-null.
// ---------------------------------------------------------------------------
__launch_bounds__(256)
__global__ void gemm_bias_kernel(const bf16* __restrict__ A,
                                 const bf16* __restrict__ W,
                                 const float* __restrict__ bias,
                                 float* __restrict__ Cf,
                                 bf16* __restrict__ Cb,
                                 int doRelu)
{
    using namespace cfg;
    __shared__ __align__(16) bf16 sA[2][BM][BK];
    __shared__ __align__(16) bf16 sW[2][BN][BK];

    const int tid  = threadIdx.x;
    const int lane = tid & 31;
    const int wave = tid >> 5;       // 0..7
    const int wm   = wave >> 2;      // 0..1 : 64-row strip
    const int wn   = wave & 3;       // 0..3 : 32-col strip
    const int m0   = blockIdx.y * BM;
    const int n0   = blockIdx.x * BN;

    const int fl      = lane & 15;          // matrix row/col within fragment
    const int halfsel = (lane >> 4) * 8;    // lanes 16..31 hold K+8 / K+24

    v8f acc[4][2];
    #pragma unroll
    for (int mt = 0; mt < 4; ++mt)
        #pragma unroll
        for (int nt = 0; nt < 2; ++nt)
            acc[mt][nt] = (v8f)(0.0f);

    const unsigned ldsA[2] = { (unsigned)(unsigned long long)&sA[0][0][0],
                               (unsigned)(unsigned long long)&sA[1][0][0] };
    const unsigned ldsW[2] = { (unsigned)(unsigned long long)&sW[0][0][0],
                               (unsigned)(unsigned long long)&sW[1][0][0] };

    // Prologue: DMA stage 0 into buffer 0
    if (wave == 0) {
        tdm_load_tile_bf16(&A[(size_t)m0 * K], ldsA[0], BM, BK, K);
        tdm_load_tile_bf16(&W[(size_t)n0 * K], ldsW[0], BN, BK, K);
        __builtin_amdgcn_s_wait_tensorcnt(0);
    }
    __syncthreads();

    int cur = 0;
    for (int k0 = 0; k0 < K; k0 += BK) {
        const int nxt = cur ^ 1;
        // DMA next stage while computing this one
        if ((k0 + BK < K) && wave == 0) {
            tdm_load_tile_bf16(&A[(size_t)m0 * K + (k0 + BK)], ldsA[nxt], BM, BK, K);
            tdm_load_tile_bf16(&W[(size_t)n0 * K + (k0 + BK)], ldsW[nxt], BN, BK, K);
        }

        #pragma unroll
        for (int kk = 0; kk < BK; kk += 32) {
            // B fragments (32x16 KxN): lane fl holds column n = row fl of W tile
            v16bf bfrag[2];
            #pragma unroll
            for (int nt = 0; nt < 2; ++nt) {
                const int nr = wn * 32 + nt * 16 + fl;
                v8bf lo = *(const v8bf*)&sW[cur][nr][kk + halfsel];
                v8bf hi = *(const v8bf*)&sW[cur][nr][kk + halfsel + 16];
                union { v16bf v; v8bf h[2]; } u;
                u.h[0] = lo; u.h[1] = hi;
                bfrag[nt] = u.v;
            }
            #pragma unroll
            for (int mt = 0; mt < 4; ++mt) {
                const int mr = wm * 64 + mt * 16 + fl;
                v8bf lo = *(const v8bf*)&sA[cur][mr][kk + halfsel];
                v8bf hi = *(const v8bf*)&sA[cur][mr][kk + halfsel + 16];
                union { v16bf v; v8bf h[2]; } u;
                u.h[0] = lo; u.h[1] = hi;
                const v16bf afrag = u.v;
                #pragma unroll
                for (int nt = 0; nt < 2; ++nt) {
                    acc[mt][nt] = __builtin_amdgcn_wmma_f32_16x16x32_bf16(
                        /*neg_a=*/false, afrag, /*neg_b=*/false, bfrag[nt],
                        /*c_mod=*/(short)0, acc[mt][nt],
                        /*reuse_a=*/false, /*reuse_b=*/false);
                }
            }
        }

        if (wave == 0) __builtin_amdgcn_s_wait_tensorcnt(0);
        __syncthreads();
        cur = nxt;
    }

    // Epilogue: C/D layout -> lane = N column, VGPR index = M row (+8 hi lanes)
    #pragma unroll
    for (int mt = 0; mt < 4; ++mt) {
        const int rowbase = m0 + wm * 64 + mt * 16 + ((lane >> 4) * 8);
        #pragma unroll
        for (int nt = 0; nt < 2; ++nt) {
            const int col = n0 + wn * 32 + nt * 16 + fl;
            const float bv = bias[col];
            #pragma unroll
            for (int r = 0; r < 8; ++r) {
                float v = acc[mt][nt][r] + bv;
                if (doRelu) v = v > 0.0f ? v : 0.0f;
                if (Cb) Cb[(size_t)(rowbase + r) * N + col] = (bf16)v;
                else    Cf[(size_t)(rowbase + r) * N + col] = v;
            }
        }
    }
}

// ---------------------------------------------------------------------------
// Per-position head-mixing attention (reference splits heads w/o transpose):
// for each (b,s): Q,K,V are 8x128; scores = QK^T/sqrt(128) (8x8),
// softmax rows, out = W@V. One thread per (b,s,i). fp32 in, bf16 out.
// ---------------------------------------------------------------------------
__launch_bounds__(256)
__global__ void head_attn_kernel(const float* __restrict__ q,
                                 const float* __restrict__ k,
                                 const float* __restrict__ v,
                                 bf16* __restrict__ out)
{
    using namespace cfg;
    const long t   = (long)blockIdx.x * 256 + threadIdx.x;  // (pos, i)
    const int  i   = (int)(t & (H - 1));
    const long pos = t >> 3;

    const float* qr = q + pos * D + (long)i * HD;
    const float* kb = k + pos * D;
    const float* vb = v + pos * D;

    float s[H];
    #pragma unroll
    for (int j = 0; j < H; ++j) {
        const f32x4* qp = (const f32x4*)qr;
        const f32x4* kp = (const f32x4*)(kb + j * HD);
        float a = 0.0f;
        #pragma unroll 4
        for (int c = 0; c < HD / 4; ++c) {
            f32x4 qa = qp[c], ka = kp[c];
            a += qa[0] * ka[0] + qa[1] * ka[1] + qa[2] * ka[2] + qa[3] * ka[3];
        }
        s[j] = a * 0.08838834764831845f;  // 1/sqrt(128)
    }
    float mx = s[0];
    #pragma unroll
    for (int j = 1; j < H; ++j) mx = fmaxf(mx, s[j]);
    float sum = 0.0f;
    #pragma unroll
    for (int j = 0; j < H; ++j) { s[j] = __expf(s[j] - mx); sum += s[j]; }
    const float inv = 1.0f / sum;

    bf16* orow = out + pos * D + (long)i * HD;
    #pragma unroll 4
    for (int c = 0; c < HD / 4; ++c) {
        f32x4 o = (f32x4)(0.0f);
        #pragma unroll
        for (int j = 0; j < H; ++j) {
            f32x4 vv = *(const f32x4*)(vb + j * HD + c * 4);
            o += vv * (s[j] * inv);
        }
        bf16x4 ob = { (bf16)o[0], (bf16)o[1], (bf16)o[2], (bf16)o[3] };
        *(bf16x4*)(orow + c * 4) = ob;
    }
}

// ---------------------------------------------------------------------------
// Launch. Inputs (setup_inputs order):
//  0 input_seq, 1 output_seq, 2 pe, then 14 (w,b) pairs:
//  enc_{q,k,v}, enc_{w1,w2}, dec_s_{q,k,v}, dec_c_{q,k,v}, dec_{w1,w2}, fc
// ---------------------------------------------------------------------------
extern "C" void kernel_launch(void* const* d_in, const int* in_sizes, int n_in,
                              void* d_out, int out_size, void* d_ws, size_t ws_size,
                              hipStream_t stream)
{
    using namespace cfg;
    const float* input_seq  = (const float*)d_in[0];
    const float* output_seq = (const float*)d_in[1];
    const float* pe         = (const float*)d_in[2];
    const float* Wt[14];
    const float* Bi[14];
    for (int i = 0; i < 14; ++i) {
        Wt[i] = (const float*)d_in[3 + 2 * i];
        Bi[i] = (const float*)d_in[4 + 2 * i];
    }
    // weight idx: 0 enc_wq 1 enc_wk 2 enc_wv 3 enc_w1 4 enc_w2
    //             5 dec_s_wq 6 dec_s_wk 7 dec_s_wv
    //             8 dec_c_wq 9 dec_c_wk 10 dec_c_wv  11 dec_w1 12 dec_w2 13 fc

    constexpr size_t ACT = (size_t)M * D;           // elements per activation
    char* p = (char*)d_ws;
    auto carve = [&](size_t bytes) {
        void* r = (void*)p;
        p += (bytes + 255) & ~(size_t)255;
        return r;
    };
    float* qf  = (float*)carve(ACT * 4);
    float* kf  = (float*)carve(ACT * 4);
    float* vf  = (float*)carve(ACT * 4);
    bf16* xbf  = (bf16*)carve(ACT * 2);     // x, later enc_out
    bf16* osbf = (bf16*)carve(ACT * 2);     // output_seq (bf16)
    bf16* abf  = (bf16*)carve(ACT * 2);
    bf16* bbf  = (bf16*)carve(ACT * 2);
    bf16* wbf[14];
    for (int i = 0; i < 14; ++i) wbf[i] = (bf16*)carve((size_t)N * K * 2);
    float* outp = (float*)d_out;

    const dim3 blk(256);
    const dim3 gemmGrid(N / BN, M / BM);            // (8, 128)

    auto GEMM = [&](const bf16* Ain, int wi, float* cf, bf16* cb, int relu) {
        gemm_bias_kernel<<<gemmGrid, blk, 0, stream>>>(Ain, wbf[wi], Bi[wi], cf, cb, relu);
    };
    auto ATTN = [&](const float* qq, const float* kk, const float* vv, bf16* oo) {
        head_attn_kernel<<<(M * H) / 256, blk, 0, stream>>>(qq, kk, vv, oo);
    };

    // one-time conversions (per call): weights + output_seq, and x = in + pe
    for (int i = 0; i < 14; ++i)
        f32_to_bf16_kernel<<<((size_t)N * K / 4) / 256, blk, 0, stream>>>(Wt[i], wbf[i]);
    f32_to_bf16_kernel<<<(ACT / 4) / 256, blk, 0, stream>>>(output_seq, osbf);
    add_pe_kernel<<<(ACT / 4) / 256, blk, 0, stream>>>(input_seq, pe, xbf);

    // ---- encoder ----
    GEMM(xbf, 0, qf, nullptr, 0);
    GEMM(xbf, 1, kf, nullptr, 0);
    GEMM(xbf, 2, vf, nullptr, 0);
    ATTN(qf, kf, vf, abf);                  // a
    GEMM(abf, 3, nullptr, bbf, 1);          // relu(a W1^T + b1)
    GEMM(bbf, 4, nullptr, xbf, 0);          // enc_out (reuses x buffer)

    // ---- decoder self-attention ----
    GEMM(osbf, 5, qf, nullptr, 0);
    GEMM(osbf, 6, kf, nullptr, 0);
    GEMM(osbf, 7, vf, nullptr, 0);
    ATTN(qf, kf, vf, abf);                  // sa

    // ---- decoder cross-attention (q from sa, k/v from enc_out) ----
    GEMM(abf, 8, qf, nullptr, 0);
    GEMM(xbf, 9, kf, nullptr, 0);
    GEMM(xbf, 10, vf, nullptr, 0);
    ATTN(qf, kf, vf, bbf);                  // ca

    // ---- decoder FFN + final fc ----
    GEMM(bbf, 11, nullptr, abf, 1);         // relu(ca W1^T + b1)
    GEMM(abf, 12, nullptr, bbf, 0);         // dec_out
    GEMM(bbf, 13, outp, nullptr, 0);        // final projection -> d_out (fp32)
}
